// mowLSTM__31920196944207
// MI455X (gfx1250) — matrix operand order repaired and checked by reading, not compiled, not run on hardware
//
#include <hip/hip_runtime.h>
#include <hip/hip_bf16.h>

// MoE-LSTM for MI455X (gfx1250), bf16 WMMA with fp32 accumulation.
//
// Phase A (prep):   pack Wi/Wh into exact v_wmma B-fragment layout (bf16),
//                   fold coef into biases, init h_rep/c/barrier.
// Phase B (parallel GEMM): xg = x_rep @ Wi_cat  -> [T*B, 2048] fp32.
// Phase C (persistent):    128 sequential steps, each [64,2048]x[2048,2048]
//                   bf16 WMMA GEMM (weights L2-resident, 8.4 MB) + LSTM cell,
//                   one atomic grid barrier per step, ping-pong h_rep.
//                   A-chunk staging uses GLOBAL_LOAD_ASYNC_TO_LDS_B128.
//
// Gate columns are permuted to n = h_unit*4 + gate so every WG owns all four
// gates of its 32 h-units -> cell update is workgroup-local.

#define TT 128
#define BB 64
#define II 256
#define HH 512
#define GG 2048
#define NWG 16

typedef __attribute__((ext_vector_type(16))) __bf16 v16bf;
typedef __attribute__((ext_vector_type(8)))  float  v8f;

struct Frag32 { uint4 lo, hi; };

__device__ inline unsigned short f2bf(float f) {
    unsigned int u = __float_as_uint(f);
    unsigned int r = u + 0x7FFFu + ((u >> 16) & 1u);   // round-to-nearest-even
    return (unsigned short)(r >> 16);
}
__device__ inline unsigned int pack2bf(float lo, float hi) {
    return (unsigned int)f2bf(lo) | ((unsigned int)f2bf(hi) << 16);
}
__device__ inline v16bf load_frag(const unsigned short* p) {
    Frag32 f;
    const uint4* q = (const uint4*)p;   // 32B-aligned by construction
    f.lo = q[0]; f.hi = q[1];
    return __builtin_bit_cast(v16bf, f);
}
__device__ inline float sigm(float x)  { return 1.0f / (1.0f + __expf(-x)); }
__device__ inline float tanhx(float x) { return 2.0f / (1.0f + __expf(-2.0f * x)) - 1.0f; }

#define WMMA_BF16(A, B, C) \
    __builtin_amdgcn_wmma_f32_16x16x32_bf16(false, (A), false, (B), (short)0, (C), false, false)

// ---------------- workspace layout (bytes) ----------------
#define OFF_WI    ((size_t)0)                         //  4 MiB  bf16 [1024 x 2048] packed
#define OFF_WH    (OFF_WI + (size_t)4*1024*1024)      //  8 MiB  bf16 [2048 x 2048] packed
#define OFF_APACK (OFF_WH + (size_t)8*1024*1024)      //  2 x 256 KiB h_rep ping-pong (bf16 frags)
#define OFF_BIAS  (OFF_APACK + (size_t)524288)        //  512 KiB fp32 [64 x 2048]
#define OFF_C     (OFF_BIAS + (size_t)524288)         //  128 KiB fp32 [64 x 512]
#define OFF_CNT   (OFF_C + (size_t)131072)            //  barrier counter
#define OFF_XG    (OFF_CNT + (size_t)256)             //  64 MiB fp32 [128*64 x 2048]
// total ~81 MiB

// ---------------- prep: pack Wi into B-fragment layout ----------------
// B frag (32x16 KxN): lane l holds column n0+(l&15), K = k0 + (j>>3)*16 + (l>>4)*8 + (j&7).
__global__ __launch_bounds__(256) void kPackWi(const float* __restrict__ Wi,
                                               unsigned short* __restrict__ wiPack) {
    int t0 = blockIdx.x * 256 + threadIdx.x;          // < 2,097,152
    int j = t0 & 15, l = (t0 >> 4) & 31;
    int rest = t0 >> 9;
    int nt = rest & 127, kt = rest >> 7;              // kt 0..31
    int k = kt * 32 + ((j >> 3) * 16) + ((l >> 4) * 8) + (j & 7); // K index (e*256+i)
    int n = nt * 16 + (l & 15);                       // permuted output column
    int e = k >> 8, i = k & 255;
    int gate = n & 3, hu = n >> 2;
    wiPack[t0] = f2bf(Wi[((size_t)(e * GG + gate * HH + hu)) * II + i]);
}

__global__ __launch_bounds__(256) void kPackWh(const float* __restrict__ Wh,
                                               unsigned short* __restrict__ whPack) {
    int t0 = blockIdx.x * 256 + threadIdx.x;          // < 4,194,304
    int j = t0 & 15, l = (t0 >> 4) & 31;
    int rest = t0 >> 9;
    int nt = rest & 127, kt = rest >> 7;              // kt 0..63
    int k = kt * 32 + ((j >> 3) * 16) + ((l >> 4) * 8) + (j & 7); // K index (e*512+h)
    int n = nt * 16 + (l & 15);
    int e = k >> 9, hh = k & 511;
    int gate = n & 3, hu = n >> 2;
    whPack[t0] = f2bf(Wh[((size_t)(e * GG + gate * HH + hu)) * HH + hh]);
}

// bias_mix, initial h_rep (A-fragment layout), c state, barrier reset
__global__ __launch_bounds__(256) void kPrepMisc(const float* __restrict__ coef,
                                                 const float* __restrict__ bi,
                                                 const float* __restrict__ bh,
                                                 const float* __restrict__ h0,
                                                 const float* __restrict__ c0,
                                                 float* __restrict__ biasMix,
                                                 unsigned short* __restrict__ aPack,
                                                 float* __restrict__ cWS,
                                                 unsigned int* __restrict__ cnt) {
    int idx = blockIdx.x * 256 + threadIdx.x;
    if (idx == 0) *cnt = 0u;
    if (idx < 131072) {                               // bias_mix[b][n] = coef.(bi+bh), permuted
        int b = idx >> 11, n = idx & 2047;
        int gate = n & 3, hu = n >> 2;
        int o = gate * HH + hu;
        float s = 0.f;
        #pragma unroll
        for (int e = 0; e < 4; ++e) s += coef[b * 4 + e] * (bi[e * GG + o] + bh[e * GG + o]);
        biasMix[idx] = s;
    } else if (idx < 262144) {                        // h_rep buffer 0 in A-frag layout
        int a0 = idx - 131072;
        int j = a0 & 15, l = (a0 >> 4) & 31;
        int rest = a0 >> 9;
        int mt = rest & 3, kt = rest >> 2;
        int b = mt * 16 + (l & 15);
        int k = kt * 32 + ((j >> 3) * 16) + ((l >> 4) * 8) + (j & 7);
        int e = k >> 9, hh = k & 511;
        aPack[a0] = f2bf(coef[b * 4 + e] * h0[b * HH + hh]);
    } else if (idx < 294912) {
        int i2 = idx - 262144;
        cWS[i2] = c0[i2];
    }
}

// ---------------- phase B: xg GEMM [8192,1024]x[1024,2048] ----------------
__global__ __launch_bounds__(256) void kInputGemm(const float* __restrict__ x,
                                                  const float* __restrict__ coef,
                                                  const unsigned short* __restrict__ wiPack,
                                                  float* __restrict__ xg) {
    __shared__ unsigned short aChunk[64 * 256];       // 32 KiB: A chunk, K=256 (one expert)
    const int tid = threadIdx.x, lane = tid & 31, w = tid >> 5;
    const int lhi = lane >> 4;
    const int wgn = blockIdx.x;                       // N-block 0..15
    const int t   = blockIdx.y;                       // timestep 0..127
    const int ntw = wgn * 8 + w;                      // N-tile 0..127

    v8f acc[4];
    #pragma unroll
    for (int m = 0; m < 4; ++m)
        #pragma unroll
        for (int r = 0; r < 8; ++r) acc[m][r] = 0.f;

    for (int e = 0; e < 4; ++e) {
        // build x_rep chunk in LDS, in exact A-fragment layout
        #pragma unroll
        for (int q = 0; q < 4; ++q) {
            int lr = tid + q * 256;                   // 0..1023 lane-rows
            int l2 = lr & 31;
            int fm = lr >> 5;
            int mt = fm & 3;
            int ktl = fm >> 2;                        // 0..7
            int b = mt * 16 + (l2 & 15);
            int lh = l2 >> 4;
            float cf = coef[b * 4 + e];
            const float* xp = x + ((size_t)(t * BB + b)) * II + ktl * 32 + lh * 8;
            float4 a0 = *(const float4*)(xp);
            float4 a1 = *(const float4*)(xp + 4);
            float4 b0 = *(const float4*)(xp + 16);
            float4 b1 = *(const float4*)(xp + 20);
            uint4 w0, w1;
            w0.x = pack2bf(a0.x * cf, a0.y * cf); w0.y = pack2bf(a0.z * cf, a0.w * cf);
            w0.z = pack2bf(a1.x * cf, a1.y * cf); w0.w = pack2bf(a1.z * cf, a1.w * cf);
            w1.x = pack2bf(b0.x * cf, b0.y * cf); w1.y = pack2bf(b0.z * cf, b0.w * cf);
            w1.z = pack2bf(b1.x * cf, b1.y * cf); w1.w = pack2bf(b1.z * cf, b1.w * cf);
            uint4* dst = (uint4*)&aChunk[lr * 16];
            dst[0] = w0; dst[1] = w1;
        }
        __syncthreads();
        #pragma unroll
        for (int ktl = 0; ktl < 8; ++ktl) {
            int kt = e * 8 + ktl;
            v16bf bf = load_frag(wiPack + ((size_t)(kt * 128 + ntw) * 32 + lane) * 16);
            // load all four A fragments first -> DS loads issue as a clause
            v16bf a0 = load_frag(&aChunk[((ktl * 4 + 0) * 32 + lane) * 16]);
            v16bf a1 = load_frag(&aChunk[((ktl * 4 + 1) * 32 + lane) * 16]);
            v16bf a2 = load_frag(&aChunk[((ktl * 4 + 2) * 32 + lane) * 16]);
            v16bf a3 = load_frag(&aChunk[((ktl * 4 + 3) * 32 + lane) * 16]);
            acc[0] = WMMA_BF16(a0, bf, acc[0]);
            acc[1] = WMMA_BF16(a1, bf, acc[1]);
            acc[2] = WMMA_BF16(a2, bf, acc[2]);
            acc[3] = WMMA_BF16(a3, bf, acc[3]);
        }
        __syncthreads();
    }
    int ncol = ntw * 16 + (lane & 15);
    #pragma unroll
    for (int m = 0; m < 4; ++m)
        #pragma unroll
        for (int r = 0; r < 8; ++r) {
            int b = m * 16 + r + lhi * 8;
            xg[((size_t)(t * BB + b)) * GG + ncol] = acc[m][r];
        }
}

// ---------------- phase C: persistent recurrence ----------------
__global__ __launch_bounds__(256) void kRecurrent(const unsigned short* __restrict__ whPack,
                                                  unsigned short* __restrict__ aPack,
                                                  const float* __restrict__ biasMix,
                                                  const float* __restrict__ xg,
                                                  const float* __restrict__ coef,
                                                  float* __restrict__ cWS,
                                                  unsigned int* __restrict__ cnt,
                                                  float* __restrict__ out) {
    __shared__ unsigned short aLds[64 * 256];         // 32 KiB: h_rep chunk K=256
    __shared__ float gLds[64 * 128];                  // 32 KiB: gate exchange
    const int tid = threadIdx.x, lane = tid & 31, w = tid >> 5;
    const int lhi = lane >> 4;
    const int wg  = blockIdx.x;                       // 0..15
    const int ntw = wg * 8 + w;
    // raw LDS byte offset for async-to-LDS
    const unsigned aLdsBase = (unsigned)(uintptr_t)(&aLds[0]);

    for (int t = 0; t < TT; ++t) {
        const unsigned short* aP = aPack + (size_t)(t & 1) * (BB * GG);
        unsigned short* aN = aPack + (size_t)((t + 1) & 1) * (BB * GG);

        v8f acc[4];
        #pragma unroll
        for (int m = 0; m < 4; ++m)
            #pragma unroll
            for (int r = 0; r < 8; ++r) acc[m][r] = 0.f;

        for (int kc = 0; kc < 8; ++kc) {
            // async copy of the 32 KiB K-chunk (contiguous in frag layout):
            // 256 threads x 8 x 16B, ASYNCcnt-tracked, no VGPR round-trip
            {
                unsigned long long gbase =
                    (unsigned long long)(uintptr_t)(aP + (size_t)kc * 16384) +
                    (unsigned long long)tid * 16u;
                unsigned lbase = aLdsBase + (unsigned)tid * 16u;
                #pragma unroll
                for (int r = 0; r < 8; ++r) {
                    unsigned lds_off = lbase + (unsigned)(r * 256 * 16);
                    unsigned long long ga = gbase + (unsigned long long)(r * 256 * 16);
                    asm volatile("global_load_async_to_lds_b128 %0, %1, off"
                                 :: "v"(lds_off), "v"(ga) : "memory");
                }
                asm volatile("s_wait_asynccnt 0x0" ::: "memory");
            }
            __syncthreads();
            #pragma unroll
            for (int ktl = 0; ktl < 8; ++ktl) {
                int kt = kc * 8 + ktl;
                v16bf bf = load_frag(whPack + ((size_t)(kt * 128 + ntw) * 32 + lane) * 16);
                v16bf a0 = load_frag(&aLds[((ktl * 4 + 0) * 32 + lane) * 16]);
                v16bf a1 = load_frag(&aLds[((ktl * 4 + 1) * 32 + lane) * 16]);
                v16bf a2 = load_frag(&aLds[((ktl * 4 + 2) * 32 + lane) * 16]);
                v16bf a3 = load_frag(&aLds[((ktl * 4 + 3) * 32 + lane) * 16]);
                acc[0] = WMMA_BF16(a0, bf, acc[0]);
                acc[1] = WMMA_BF16(a1, bf, acc[1]);
                acc[2] = WMMA_BF16(a2, bf, acc[2]);
                acc[3] = WMMA_BF16(a3, bf, acc[3]);
            }
            __syncthreads();
        }

        // dump hg tile [64 x 128] to LDS for the cell update
        #pragma unroll
        for (int m = 0; m < 4; ++m)
            #pragma unroll
            for (int r = 0; r < 8; ++r)
                gLds[(m * 16 + r + lhi * 8) * 128 + (w * 16 + (lane & 15))] = acc[m][r];
        __syncthreads();

        // LSTM cell: this WG owns h-units [wg*32, wg*32+32), all 4 gates local
        for (int p = tid; p < 2048; p += 256) {
            int b = p >> 5;
            int hl = p & 31;
            int hu = wg * 32 + hl;
            int nb = wg * 128 + hl * 4;
            size_t xgi = ((size_t)(t * BB + b)) * GG + nb;
            size_t bsi = (size_t)b * GG + nb;
            float g0 = gLds[b * 128 + hl * 4 + 0] + xg[xgi + 0] + biasMix[bsi + 0];
            float g1 = gLds[b * 128 + hl * 4 + 1] + xg[xgi + 1] + biasMix[bsi + 1];
            float g2 = gLds[b * 128 + hl * 4 + 2] + xg[xgi + 2] + biasMix[bsi + 2];
            float g3 = gLds[b * 128 + hl * 4 + 3] + xg[xgi + 3] + biasMix[bsi + 3];
            float ig = sigm(g0), fg = sigm(g1), cg = tanhx(g2), og = sigm(g3);
            float co = cWS[b * HH + hu];
            float cn = fg * co + ig * cg;
            cWS[b * HH + hu] = cn;
            float hn = og * tanhx(cn);
            out[((size_t)(t * BB + b)) * HH + hu] = hn;
            if (t == TT - 1) {
                out[(size_t)TT * BB * HH + (size_t)b * HH + hu] = hn;
                out[(size_t)TT * BB * HH + (size_t)BB * HH + (size_t)b * HH + hu] = cn;
            }
            // write next-step h_rep into A-fragment layout (disjoint K rows per WG)
            #pragma unroll
            for (int e = 0; e < 4; ++e) {
                int k = e * HH + hu;
                int kt = k >> 5, r = k & 31;
                int lh = (r >> 3) & 1;
                int j = ((r >> 4) << 3) | (r & 7);
                int mt = b >> 4;
                int ln = (lh << 4) | (b & 15);
                aN[((size_t)(kt * 4 + mt) * 32 + ln) * 16 + j] = f2bf(coef[b * 4 + e] * hn);
            }
        }

        // one grid barrier per step (ping-pong h_rep removes the second)
        __threadfence();
        __syncthreads();
        if (tid == 0) {
            __hip_atomic_fetch_add(cnt, 1u, __ATOMIC_RELEASE, __HIP_MEMORY_SCOPE_AGENT);
            unsigned target = (unsigned)NWG * (unsigned)(t + 1);
            while (__hip_atomic_load(cnt, __ATOMIC_ACQUIRE, __HIP_MEMORY_SCOPE_AGENT) < target)
                __builtin_amdgcn_s_sleep(1);
        }
        __syncthreads();
        __threadfence();
    }
}

extern "C" void kernel_launch(void* const* d_in, const int* in_sizes, int n_in,
                              void* d_out, int out_size, void* d_ws, size_t ws_size,
                              hipStream_t stream) {
    (void)in_sizes; (void)n_in; (void)out_size; (void)ws_size; // needs ~81 MiB ws
    const float* x    = (const float*)d_in[0];
    const float* h0   = (const float*)d_in[1];
    const float* c0   = (const float*)d_in[2];
    const float* coef = (const float*)d_in[3];
    const float* Wi   = (const float*)d_in[4];
    const float* bi   = (const float*)d_in[5];
    const float* Wh   = (const float*)d_in[6];
    const float* bh   = (const float*)d_in[7];
    float* out = (float*)d_out;
    char* ws = (char*)d_ws;

    unsigned short* wiPack = (unsigned short*)(ws + OFF_WI);
    unsigned short* whPack = (unsigned short*)(ws + OFF_WH);
    unsigned short* aPack  = (unsigned short*)(ws + OFF_APACK);
    float*          biasM  = (float*)(ws + OFF_BIAS);
    float*          cWS    = (float*)(ws + OFF_C);
    unsigned int*   cnt    = (unsigned int*)(ws + OFF_CNT);
    float*          xg     = (float*)(ws + OFF_XG);

    kPackWi<<<8192, 256, 0, stream>>>(Wi, wiPack);
    kPackWh<<<16384, 256, 0, stream>>>(Wh, whPack);
    kPrepMisc<<<1152, 256, 0, stream>>>(coef, bi, bh, h0, c0, biasM, aPack, cWS, cnt);
    kInputGemm<<<dim3(16, TT), 256, 0, stream>>>(x, coef, wiPack, xg);
    kRecurrent<<<NWG, 256, 0, stream>>>(whPack, aPack, biasM, xg, coef, cWS, cnt, out);
}
